// GNN_61589831024794
// MI455X (gfx1250) — matrix operand order repaired and checked by reading, not compiled, not run on hardware
//
#include <hip/hip_runtime.h>
#include <hip/hip_bf16.h>

#define GNN_N 100000
#define GNN_E 640000
#define GNN_D 128

typedef __attribute__((ext_vector_type(16))) __bf16 v16bf;
typedef __attribute__((ext_vector_type(2)))  __bf16 bf16x2;
typedef __attribute__((ext_vector_type(8)))  float  v8f;

#define WT_STRIDE 136   // padded bf16 row stride (272B, 16B-aligned, no bank conflicts)

// ---------------------------------------------------------------------------
// Edge aggregation: dst[row[e], :] += src[col[e], :]  (one wave32 per edge)
// Lane: one float4 gather (coalesced 512B per edge) + 4 f32 scatter-adds.
// ---------------------------------------------------------------------------
__global__ __launch_bounds__(256) void gnn_agg_kernel(
    const float* __restrict__ src,
    const long long* __restrict__ edge_index,   // [2, E] int64
    float* __restrict__ dst,
    int n_edges)
{
  int gwave = (int)((blockIdx.x * 256u + threadIdx.x) >> 5);
  int lane  = threadIdx.x & 31;
  if (gwave >= n_edges) return;

  unsigned r = (unsigned)edge_index[gwave];                    // destination
  unsigned c = (unsigned)edge_index[(size_t)n_edges + gwave];  // source

  const float4* sp = (const float4*)(src + (size_t)c * GNN_D);
  float4 v = sp[lane];

  float* dp = dst + (size_t)r * GNN_D + lane * 4;
  unsafeAtomicAdd(dp + 0, v.x);
  unsafeAtomicAdd(dp + 1, v.y);
  unsafeAtomicAdd(dp + 2, v.z);
  unsafeAtomicAdd(dp + 3, v.w);
}

// ---------------------------------------------------------------------------
// Per-wave tile loop: 8 col-tiles x 4 K-steps of v_wmma_f32_16x16x32_bf16,
// epilogue fused. FULL=true -> fully unguarded straight-line path.
// ---------------------------------------------------------------------------
template<bool RELU, bool RES, bool FULL>
__device__ __forceinline__ void gnn_gemm_tiles(
    const __bf16* __restrict__ Wt, const v16bf afrag[4],
    const float bvs[8], float* __restrict__ Out,
    const float* __restrict__ Res,
    int lo, int hi, int mbase, unsigned obase, int n_rows)
{
  #pragma unroll
  for (int nt = 0; nt < 8; ++nt) {
    int n0 = nt * 16;
    v8f acc = {};

    #pragma unroll
    for (int kt = 0; kt < 4; ++kt) {
      // B layout (32x16, 16-bit): lane (lo,hi) holds column n0+lo,
      // K = {kt*32+8*hi..+7},{kt*32+16+8*hi..+7} (2x ds_load_b128)
      typedef __attribute__((ext_vector_type(8))) __bf16 v8bf_t;
      const __bf16* wp = &Wt[(n0 + lo) * WT_STRIDE + kt * 32 + 8 * hi];
      v8bf_t b0 = *(const v8bf_t*)wp;
      v8bf_t b1 = *(const v8bf_t*)(wp + 16);
      v16bf b = __builtin_shufflevector(b0, b1, 0, 1, 2, 3, 4, 5, 6, 7, 8, 9, 10, 11, 12, 13, 14, 15);
      acc = __builtin_amdgcn_wmma_f32_16x16x32_bf16(
          false, afrag[kt], false, b, (short)0, acc, false, false);
    }

    float bv = bvs[nt];
    unsigned off0 = obase + (unsigned)(n0 + lo);

    #pragma unroll
    for (int r = 0; r < 8; ++r) {
      if (FULL || (mbase + r) < n_rows) {
        float v = acc[r] + bv;
        if (RELU) v = fmaxf(v, 0.0f);
        if (RES)  v += Res[off0 + (unsigned)r * 128u];
        Out[off0 + (unsigned)r * 128u] = v;
      }
    }
  }
}

// ---------------------------------------------------------------------------
// GEMM: Out[N,128] = act(A[N,128] @ W[128,128] + bias) (+ Res)
// Block = 256 threads = 8 waves; each block computes 128 rows x 128 cols.
// W staged to LDS as bf16, transposed, padded row stride 136 so the 32B
// B-fragment reads are bank-conflict-free.
// ---------------------------------------------------------------------------
template<bool RELU, bool RES>
__global__ __launch_bounds__(256) void gnn_gemm_kernel(
    const float* __restrict__ A,
    const float* __restrict__ W,      // row-major [k][n], f32
    const float* __restrict__ bias,   // [128]
    float* __restrict__ Out,
    const float* __restrict__ Res,    // residual (only read if RES)
    int n_rows)
{
  __shared__ __align__(16) __bf16 Wt[GNN_D * WT_STRIDE];  // ~34 KB

  int tid = threadIdx.x;

  // ---- Stage weights: W[k][n] f32 -> Wt[n][k] bf16, packed b32 stores ----
  // Each iteration handles two consecutive k rows so (k, k+1) pack into one
  // bf16x2 dword store (halves DS-op count vs per-element b16 stores).
  {
    const float4* W4 = (const float4*)W;
    bf16x2* wt2 = (bf16x2*)Wt;
    #pragma unroll
    for (int j = 0; j < 8; ++j) {
      int p  = tid + 256 * j;          // 0..2047 pair-of-rows x col-group
      int kp = p >> 5;                 // k pair index: rows 2kp, 2kp+1
      int cg = p & 31;                 // float4 column group
      int n4 = cg * 4;
      float4 wa = W4[(2 * kp) * 32 + cg];
      float4 wb = W4[(2 * kp + 1) * 32 + cg];
      bf16x2 p0; p0[0] = (__bf16)wa.x; p0[1] = (__bf16)wb.x;
      bf16x2 p1; p1[0] = (__bf16)wa.y; p1[1] = (__bf16)wb.y;
      bf16x2 p2; p2[0] = (__bf16)wa.z; p2[1] = (__bf16)wb.z;
      bf16x2 p3; p3[0] = (__bf16)wa.w; p3[1] = (__bf16)wb.w;
      wt2[(n4 + 0) * (WT_STRIDE / 2) + kp] = p0;
      wt2[(n4 + 1) * (WT_STRIDE / 2) + kp] = p1;
      wt2[(n4 + 2) * (WT_STRIDE / 2) + kp] = p2;
      wt2[(n4 + 3) * (WT_STRIDE / 2) + kp] = p3;
    }
  }
  __syncthreads();

  int wave = tid >> 5;
  int lane = tid & 31;
  int hi   = lane >> 4;                // lane half: 0 or 1
  int lo   = lane & 15;

  int row0 = blockIdx.x * 128 + wave * 16;       // this wave's 16-row tile
  bool full = (row0 + 16 <= n_rows);             // wave-uniform

  // ---- Preload bias (8 per-lane values, one load clause) ----
  float bvs[8];
  #pragma unroll
  for (int nt = 0; nt < 8; ++nt) bvs[nt] = bias[nt * 16 + lo];

  // ---- A fragments for the 4 K-steps (ISA 16-bit A layout) ----
  // lane (lo,hi) holds row m = row0+lo; K runs {kb..kb+7},{kb+16..kb+23},
  // kb = kt*32 + 8*hi.
  v16bf afrag[4];
  {
    int m = row0 + lo;
    const float* arow = A + (unsigned)m * 128u;
    bool mok = full || (m < n_rows);
    #pragma unroll
    for (int kt = 0; kt < 4; ++kt) {
      v16bf a;
      if (mok) {
        int kb = kt * 32 + 8 * hi;
        float4 x0 = *(const float4*)(arow + kb);
        float4 x1 = *(const float4*)(arow + kb + 4);
        float4 x2 = *(const float4*)(arow + kb + 16);
        float4 x3 = *(const float4*)(arow + kb + 20);
        a[0]  = (__bf16)x0.x; a[1]  = (__bf16)x0.y; a[2]  = (__bf16)x0.z; a[3]  = (__bf16)x0.w;
        a[4]  = (__bf16)x1.x; a[5]  = (__bf16)x1.y; a[6]  = (__bf16)x1.z; a[7]  = (__bf16)x1.w;
        a[8]  = (__bf16)x2.x; a[9]  = (__bf16)x2.y; a[10] = (__bf16)x2.z; a[11] = (__bf16)x2.w;
        a[12] = (__bf16)x3.x; a[13] = (__bf16)x3.y; a[14] = (__bf16)x3.z; a[15] = (__bf16)x3.w;
      } else {
        #pragma unroll
        for (int jj = 0; jj < 16; ++jj) a[jj] = (__bf16)0.0f;
      }
      afrag[kt] = a;
    }
  }

  int mbase = row0 + 8 * hi;                 // D layout: vgpr r -> m=mbase+r
  unsigned obase = (unsigned)mbase * 128u;

  // ---- Branch ONCE per wave: straight-line fast path for full tiles ----
  if (full) {
    gnn_gemm_tiles<RELU, RES, true>(Wt, afrag, bvs, Out, Res,
                                    lo, hi, mbase, obase, n_rows);
  } else {
    gnn_gemm_tiles<RELU, RES, false>(Wt, afrag, bvs, Out, Res,
                                     lo, hi, mbase, obase, n_rows);
  }
}

// ---------------------------------------------------------------------------
// Launch: agg0 -> gemm0(relu) -> agg1 -> gemm1(relu + residual) -> proj
// ---------------------------------------------------------------------------
extern "C" void kernel_launch(void* const* d_in, const int* in_sizes, int n_in,
                              void* d_out, int out_size, void* d_ws, size_t ws_size,
                              hipStream_t stream) {
  const float*     x  = (const float*)d_in[0];
  const long long* ei = (const long long*)d_in[1];
  const float*     W0 = (const float*)d_in[2];
  const float*     b0 = (const float*)d_in[3];
  const float*     W1 = (const float*)d_in[4];
  const float*     b1 = (const float*)d_in[5];
  const float*     Wp = (const float*)d_in[6];
  const float*     bp = (const float*)d_in[7];
  float* out = (float*)d_out;

  const size_t feat_elems = (size_t)GNN_N * GNN_D;
  float* bufA = (float*)d_ws;            // agg0 / agg1
  float* bufB = bufA + feat_elems;       // h / h2

  const int agg_blocks  = GNN_E / 8;                 // 8 edges (waves) / block
  const int gemm_blocks = (GNN_N + 127) / 128;

  // Layer 0: agg0 = scatter_add(x), h = relu(agg0 @ W0 + b0)
  hipMemsetAsync(bufA, 0, feat_elems * sizeof(float), stream);
  gnn_agg_kernel<<<agg_blocks, 256, 0, stream>>>(x, ei, bufA, GNN_E);
  gnn_gemm_kernel<true, false><<<gemm_blocks, 256, 0, stream>>>(
      bufA, W0, b0, bufB, nullptr, GNN_N);

  // Layer 1: agg1 = scatter_add(h), h2 = relu(agg1 @ W1 + b1) + agg1
  hipMemsetAsync(bufA, 0, feat_elems * sizeof(float), stream);
  gnn_agg_kernel<<<agg_blocks, 256, 0, stream>>>(bufB, ei, bufA, GNN_E);
  gnn_gemm_kernel<true, true><<<gemm_blocks, 256, 0, stream>>>(
      bufA, W1, b1, bufB, bufA, GNN_N);

  // Projection: out = h2 @ Wp + bp
  gnn_gemm_kernel<false, false><<<gemm_blocks, 256, 0, stream>>>(
      bufB, Wp, bp, out, nullptr, GNN_N);
}
